// GeometricMambaLayer_4827543241081
// MI455X (gfx1250) — matrix operand order repaired and checked by reading, not compiled, not run on hardware
//
#include <hip/hip_runtime.h>

#define D_MODEL   1024
#define D_INNER   2048
#define NHEADS    32
#define HEADDIM   64
#define DSTATE    128
#define DCONV     4
#define CONV_DIM  (D_INNER + 2*DSTATE)              // 2304
#define PROJ_DIM  (2*D_INNER + 2*DSTATE + NHEADS)   // 4384
#define PROJ_PAD  4480                              // 35*128, padded for GEMM
#define BATCH     4
#define SEQLEN    2048
#define MROWS     (BATCH*SEQLEN)                    // 8192

typedef __attribute__((ext_vector_type(16))) __bf16        v16bf;
typedef __attribute__((ext_vector_type(8)))  float         v8f;
typedef __attribute__((ext_vector_type(4)))  unsigned int  u32x4;

union AB16 { u32x4 u[2]; v16bf v; };

// CDNA5 async global->LDS copies (ASYNCcnt), if the toolchain exposes them.
#if defined(__has_builtin)
#  if __has_builtin(__builtin_amdgcn_global_load_async_to_lds_b128) && \
      __has_builtin(__builtin_amdgcn_s_wait_asynccnt)
#    define HAVE_ASYNC_LDS 1
#  else
#    define HAVE_ASYNC_LDS 0
#  endif
#  if __has_builtin(__builtin_amdgcn_tensor_load_to_lds) && \
      __has_builtin(__builtin_amdgcn_s_wait_tensorcnt)
#    define HAVE_TDM 1
#  else
#    define HAVE_TDM 0
#  endif
#else
#  define HAVE_ASYNC_LDS 0
#  define HAVE_TDM 0
#endif

#if HAVE_ASYNC_LDS
typedef __attribute__((ext_vector_type(4))) int v4i;
typedef __attribute__((address_space(1))) v4i* gptr_v4i;   // global
typedef __attribute__((address_space(3))) v4i* sptr_v4i;   // LDS
#endif

#if HAVE_TDM
typedef __attribute__((ext_vector_type(4))) unsigned int ui32x4;
typedef __attribute__((ext_vector_type(8))) int          i32x8;
typedef __attribute__((ext_vector_type(4))) int          i32x4;
__device__ __forceinline__ unsigned lds_byte_off(const void* p) {
  return (unsigned)(unsigned long long)
      (__attribute__((address_space(3))) const void*)p;
}
#endif

__device__ __forceinline__ unsigned short f2bf(float f) {
  unsigned int u = __builtin_bit_cast(unsigned int, f);
  u += 0x7FFFu + ((u >> 16) & 1u);
  return (unsigned short)(u >> 16);
}

// ---------------------------------------------------------------- cvt f32->bf16
__global__ void k_cvt_bf16(const float* __restrict__ in,
                           unsigned short* __restrict__ out, long n) {
  long i = (long)blockIdx.x * blockDim.x + threadIdx.x;
  if (i < n) out[i] = f2bf(in[i]);
}

// ------------------------------------------- cvt + transpose (+pad) weights
// in: [K][N] f32 row-major  ->  out: [Npad][K] bf16 row-major (rows >= N are 0)
__global__ void k_cvt_bf16_T(const float* __restrict__ in,
                             unsigned short* __restrict__ out,
                             int K, int N, int Npad) {
  long i = (long)blockIdx.x * 256 + threadIdx.x;
  if (i >= (long)K * Npad) return;
  int n = (int)(i % Npad);
  int k = (int)(i / Npad);
  out[(long)n * K + k] = (n < N) ? f2bf(in[(long)k * N + n]) : (unsigned short)0;
}

// ---------------------------------------------------------------- bf16 WMMA GEMM
// C[M,N](f32) = A[M,K](bf16 row-major) @ BT[N,K](bf16, pre-transposed)^T
// Requires: M % 128 == 0, N % 128 == 0, K % 32 == 0.
// REVA: read A rows per-batch time-reversed; REVC: write C rows time-reversed;
// ACC: C += ; EPI==1: C = resid + 0.1*tanh(acc + bias[col])
template<int REVA, int REVC, int ACC, int EPI>
__global__ __launch_bounds__(256)
void k_gemm_bf16(const unsigned short* __restrict__ A,
                 const unsigned short* __restrict__ BT,
                 float* __restrict__ C,
                 int M, int N, int K, int L,
                 const float* __restrict__ resid,
                 const float* __restrict__ bias)
{
  __shared__ __align__(16) unsigned short As[2][128][40];
  __shared__ __align__(16) unsigned short Bs[2][128][40];   // [n][k]

  const int tid  = threadIdx.x;
  const int lane = tid & 31;
  const int wave = tid >> 5;
  const int wr   = wave >> 1;     // 0..3 : 32 rows of M each
  const int wc   = wave & 1;      // 0..1 : 64 cols of N each
  const int l15  = lane & 15;
  const int lh   = lane >> 4;     // 0/1

  const long mBase = (long)blockIdx.y * 128;
  const long nBase = (long)blockIdx.x * 128;

  // staging: thread moves 16 contiguous bf16 of A row rr (and BT row rr w/o TDM)
  const int rr  = tid >> 1;          // 0..127
  const int seg = (tid & 1) << 4;    // 0 or 16
  long am = mBase + rr;
  if (REVA) am = (am / L) * L + (L - 1 - (am % L));
  const unsigned short* Agp = A  + am * (long)K + seg;
  const unsigned short* Bgp = BT + (nBase + rr) * (long)K + seg;

  v8f acc[2][4];
  const v8f vzero = {0.f,0.f,0.f,0.f,0.f,0.f,0.f,0.f};
#pragma unroll
  for (int r = 0; r < 2; ++r)
#pragma unroll
    for (int c = 0; c < 4; ++c) acc[r][c] = vzero;

  const int nk = K >> 5;

  auto stage = [&](int kt, int pb) {
    const unsigned short* ga = Agp + (kt << 5);
    // ---- A tile: per-thread async copies (handles REVA row remap)
#if HAVE_ASYNC_LDS
    __builtin_amdgcn_global_load_async_to_lds_b128(
        (gptr_v4i)ga, (sptr_v4i)&As[pb][rr][seg], 0, 0);
    __builtin_amdgcn_global_load_async_to_lds_b128(
        (gptr_v4i)ga, (sptr_v4i)&As[pb][rr][seg], 16, 0);
#else
    const u32x4* pa = (const u32x4*)ga;
    u32x4 a0 = pa[0], a1 = pa[1];
    *(u32x4*)&As[pb][rr][seg]     = a0;
    *(u32x4*)&As[pb][rr][seg + 8] = a1;
#endif
    // ---- B tile: TDM 2D tile load (128 rows x 32 bf16, row stride K),
    //      LDS pad: 16 DWORDs data + 4 DWORDs pad == 80B row pitch of Bs.
#if HAVE_TDM
    if (wave == 0) {
      const unsigned long long gaddr =
          (unsigned long long)(const void*)(BT + nBase * (long)K + ((long)kt << 5));
      const unsigned ldsa = lds_byte_off(&Bs[pb][0][0]);
      ui32x4 g0 = { 1u,                                    // count=1 (valid D#)
                    ldsa,                                  // lds_addr
                    (unsigned)(gaddr & 0xFFFFFFFFu),       // global_addr lo
                    (unsigned)((gaddr >> 32) & 0x01FFFFFFu) | (2u << 30) }; // hi + type=2
      const unsigned td1 = 0x100000u;                      // tensor_dim1 (large)
      const unsigned w0  = (1u << 16)    // data_size = 2 bytes
                         | (1u << 20)    // pad_enable
                         | (3u << 22)    // pad_interval = 16 DWORDs
                         | (3u << 25);   // pad_amount   = 4 DWORDs
      i32x8 g1 = { (int)w0,
                   (int)(((unsigned)K & 0xFFFFu) << 16),                     // dim0 lo16 @bits63:48
                   (int)((((unsigned)K >> 16) & 0xFFFFu) | ((td1 & 0xFFFFu) << 16)),
                   (int)(((td1 >> 16) & 0xFFFFu) | (32u << 16)),             // tile_dim0=32
                   (int)128,                                                 // tile_dim1=128
                   (int)K,                                                   // dim0_stride lo32
                   0, 0 };
      i32x4 gz  = {0, 0, 0, 0};
      i32x8 gz8 = {0, 0, 0, 0, 0, 0, 0, 0};
      __builtin_amdgcn_tensor_load_to_lds(g0, g1, gz, gz, gz8, 0);
    }
#elif HAVE_ASYNC_LDS
    const unsigned short* gb = Bgp + (kt << 5);
    __builtin_amdgcn_global_load_async_to_lds_b128(
        (gptr_v4i)gb, (sptr_v4i)&Bs[pb][rr][seg], 0, 0);
    __builtin_amdgcn_global_load_async_to_lds_b128(
        (gptr_v4i)gb, (sptr_v4i)&Bs[pb][rr][seg], 16, 0);
#else
    const unsigned short* gb = Bgp + (kt << 5);
    const u32x4* pb4 = (const u32x4*)gb;
    u32x4 b0 = pb4[0], b1 = pb4[1];
    *(u32x4*)&Bs[pb][rr][seg]     = b0;
    *(u32x4*)&Bs[pb][rr][seg + 8] = b1;
#endif
  };

  stage(0, 0);

  for (int kt = 0; kt < nk; ++kt) {
    const int pb = kt & 1;
#if HAVE_ASYNC_LDS
    __builtin_amdgcn_s_wait_asynccnt(0);
#endif
#if HAVE_TDM
    __builtin_amdgcn_s_wait_tensorcnt(0);
#endif
    __syncthreads();
    if (kt + 1 < nk) {
      __builtin_prefetch(Agp + ((kt + 2) << 5), 0, 0);
      __builtin_prefetch(Bgp + ((kt + 2) << 5), 0, 0);
      stage(kt + 1, pb ^ 1);
    }

    // fragments per ISA wave32 layouts (CDNA5 ISA 7.12.2)
    v16bf af[2], bfrag[4];
#pragma unroll
    for (int r = 0; r < 2; ++r) {
      const int row = wr * 32 + r * 16 + l15;
      const int kh  = lh * 8;                 // lanes 16-31: K 8..15 / 24..31
      AB16 u;
      u.u[0] = *(const u32x4*)&As[pb][row][kh];
      u.u[1] = *(const u32x4*)&As[pb][row][16 + kh];
      af[r] = u.v;
    }
#pragma unroll
    for (int c = 0; c < 4; ++c) {
      const int n  = wc * 64 + c * 16 + l15;
      const int kb = lh * 8;                  // lanes 16-31: K 8..15 / 24..31
      AB16 u;
      u.u[0] = *(const u32x4*)&Bs[pb][n][kb];
      u.u[1] = *(const u32x4*)&Bs[pb][n][16 + kb];
      bfrag[c] = u.v;
    }
#pragma unroll
    for (int r = 0; r < 2; ++r)
#pragma unroll
      for (int c = 0; c < 4; ++c)
        acc[r][c] = __builtin_amdgcn_wmma_f32_16x16x32_bf16(
            false, af[r], false, bfrag[c], (short)0, acc[r][c], false, false);
  }

  // Epilogue: C/D layout -> lane l15 = N col, VGPR v = M row (+8 for lanes 16..31)
#pragma unroll
  for (int r = 0; r < 2; ++r) {
#pragma unroll
    for (int c = 0; c < 4; ++c) {
      const long col = nBase + wc * 64 + c * 16 + l15;
#pragma unroll
      for (int v = 0; v < 8; ++v) {
        long grow = mBase + wr * 32 + r * 16 + lh * 8 + v;
        long orow = REVC ? (grow / L) * L + (L - 1 - (grow % L)) : grow;
        long idx  = orow * (long)N + col;
        float val = acc[r][c][v];
        if (EPI) {
          C[idx] = resid[idx] + 0.1f * tanhf(val + bias[col]);
        } else if (ACC) {
          C[idx] += val;
        } else {
          C[idx] = val;
        }
      }
    }
  }
}

// ---------------------------------------------------------------- depthwise conv + SiLU
__global__ void k_conv_silu(const float* __restrict__ proj,
                            const float* __restrict__ w,
                            const float* __restrict__ wb,
                            float* __restrict__ out)
{
  long i = (long)blockIdx.x * 256 + threadIdx.x;
  if (i >= (long)MROWS * CONV_DIM) return;
  int  c = (int)(i % CONV_DIM);
  long m = i / CONV_DIM;
  int  l = (int)(m % SEQLEN);
  long b = m / SEQLEN;
  float s = wb[c];
#pragma unroll
  for (int j = 0; j < DCONV; ++j) {
    int ls = l - (DCONV - 1) + j;
    if (ls >= 0)
      s += w[j * CONV_DIM + c] *
           proj[(b * SEQLEN + ls) * (long)PROJ_PAD + D_INNER + c];
  }
  out[i] = s / (1.f + __expf(-s));           // silu
}

// ------------------------------------------- dt = softplus, dA = exp(-dt*exp(Alog))
__global__ void k_dt_dA(const float* __restrict__ proj,
                        const float* __restrict__ dtb,
                        const float* __restrict__ Alog,
                        float* __restrict__ dtv, float* __restrict__ dAv)
{
  long i = (long)blockIdx.x * 256 + threadIdx.x;
  if (i >= (long)MROWS * NHEADS) return;
  int  h = (int)(i & (NHEADS - 1));
  long m = i >> 5;
  float xv = proj[m * (long)PROJ_PAD + (D_INNER + CONV_DIM) + h] + dtb[h];
  float sp = (xv > 20.f) ? xv : log1pf(__expf(xv));
  dtv[i] = sp;
  dAv[i] = __expf(-sp * __expf(Alog[h]));
}

// ---------------------------------------------------------------- sequential SSD scan
// one block per (b,h); state h[64][128] in regs: thread t owns p=t>>2, n in [(t&3)*32,+32)
__global__ __launch_bounds__(256)
void k_scan(const float* __restrict__ conv, const float* __restrict__ proj,
            const float* __restrict__ dtv,  const float* __restrict__ dAv,
            const float* __restrict__ Dh,   unsigned short* __restrict__ yout)
{
  __shared__ float sb[2][320];   // [x(64) | B(128) | C(128)] double-buffered
  const int bid = blockIdx.x;
  const int b   = bid >> 5;
  const int h   = bid & 31;
  const int tid = threadIdx.x;
  const int p   = tid >> 2;
  const int q   = tid & 3;

  float hreg[32];
#pragma unroll
  for (int j = 0; j < 32; ++j) hreg[j] = 0.f;
  const float dcoef = Dh[h];
  const long  mbase = (long)b * SEQLEN;

  for (int l = 0; l < SEQLEN; ++l) {
    const long m = mbase + l;
    const float* row = conv + m * (long)CONV_DIM;
    float* s = sb[l & 1];
    for (int i = tid; i < 320; i += 256) {
      float v;
      if (i < 64)       v = row[h * 64 + i];
      else if (i < 192) v = row[D_INNER + (i - 64)];
      else              v = row[D_INNER + DSTATE + (i - 192)];
      s[i] = v;
    }
    __syncthreads();

    const float a    = dAv[m * NHEADS + h];
    const float dt   = dtv[m * NHEADS + h];
    const float xp   = s[p];
    const float coef = dt * xp;
    const float* Bv  = s + 64  + q * 32;
    const float* Cv  = s + 192 + q * 32;
    float acc = 0.f;
#pragma unroll
    for (int j = 0; j < 32; ++j) {
      hreg[j] = a * hreg[j] + coef * Bv[j];
      acc += hreg[j] * Cv[j];
    }
    acc += __shfl_xor(acc, 1, 32);
    acc += __shfl_xor(acc, 2, 32);
    if (q == 0) {
      float z = proj[m * (long)PROJ_PAD + h * 64 + p];
      float y = (acc + dcoef * xp) * (z / (1.f + __expf(-z)));
      yout[m * (long)D_INNER + h * 64 + p] = f2bf(y);
    }
  }
}

// ---------------------------------------------------------------- host launcher
extern "C" void kernel_launch(void* const* d_in, const int* in_sizes, int n_in,
                              void* d_out, int out_size, void* d_ws, size_t ws_size,
                              hipStream_t stream)
{
  (void)in_sizes; (void)n_in; (void)out_size; (void)ws_size;
  const float* x     = (const float*)d_in[0];
  const float* Wi[2] = {(const float*)d_in[1],  (const float*)d_in[8]};
  const float* cw[2] = {(const float*)d_in[2],  (const float*)d_in[9]};
  const float* cb[2] = {(const float*)d_in[3],  (const float*)d_in[10]};
  const float* Al[2] = {(const float*)d_in[4],  (const float*)d_in[11]};
  const float* Dh[2] = {(const float*)d_in[5],  (const float*)d_in[12]};
  const float* db[2] = {(const float*)d_in[6],  (const float*)d_in[13]};
  const float* Wo[2] = {(const float*)d_in[7],  (const float*)d_in[14]};
  const float* Wm    = (const float*)d_in[15];
  const float* bm    = (const float*)d_in[16];
  float* out = (float*)d_out;

  size_t off = 0;
  char* base = (char*)d_ws;
  auto carve = [&](size_t bytes) -> void* {
    void* p = base + off;
    off += (bytes + 255) & ~(size_t)255;
    return p;
  };
  float*          proj    = (float*)carve(sizeof(float)*(size_t)MROWS*PROJ_PAD);
  float*          convact = (float*)carve(sizeof(float)*(size_t)MROWS*CONV_DIM);
  float*          xm      = (float*)carve(sizeof(float)*(size_t)MROWS*D_MODEL);
  float*          dtv     = (float*)carve(sizeof(float)*(size_t)MROWS*NHEADS);
  float*          dAv     = (float*)carve(sizeof(float)*(size_t)MROWS*NHEADS);
  unsigned short* xbf     = (unsigned short*)carve(2ULL*MROWS*D_MODEL);
  unsigned short* ybf     = (unsigned short*)carve(2ULL*MROWS*D_INNER);
  unsigned short* xmbf    = (unsigned short*)carve(2ULL*MROWS*D_MODEL);
  unsigned short* WiT     = (unsigned short*)carve(2ULL*PROJ_PAD*D_MODEL);
  unsigned short* WoT     = (unsigned short*)carve(2ULL*D_MODEL*D_INNER);
  unsigned short* WmT     = (unsigned short*)carve(2ULL*D_MODEL*D_MODEL);

  auto cvt = [&](const float* src, unsigned short* dst, long n) {
    k_cvt_bf16<<<(unsigned)((n + 255) / 256), 256, 0, stream>>>(src, dst, n);
  };
  auto cvtT = [&](const float* src, unsigned short* dst, int K, int N, int Npad) {
    long n = (long)K * Npad;
    k_cvt_bf16_T<<<(unsigned)((n + 255) / 256), 256, 0, stream>>>(src, dst, K, N, Npad);
  };

  cvt(x, xbf, (long)MROWS * D_MODEL);
  cvtT(Wm, WmT, D_MODEL, D_MODEL, D_MODEL);

  dim3 blk(256);
  for (int d = 0; d < 2; ++d) {
    cvtT(Wi[d], WiT, D_MODEL, PROJ_DIM, PROJ_PAD);
    cvtT(Wo[d], WoT, D_INNER, D_MODEL, D_MODEL);

    dim3 g1(PROJ_PAD / 128, MROWS / 128);
    if (d == 0)
      k_gemm_bf16<0,0,0,0><<<g1, blk, 0, stream>>>(xbf, WiT, proj,
          MROWS, PROJ_PAD, D_MODEL, SEQLEN, nullptr, nullptr);
    else
      k_gemm_bf16<1,0,0,0><<<g1, blk, 0, stream>>>(xbf, WiT, proj,
          MROWS, PROJ_PAD, D_MODEL, SEQLEN, nullptr, nullptr);

    long ctotal = (long)MROWS * CONV_DIM;
    k_conv_silu<<<(unsigned)((ctotal + 255) / 256), blk, 0, stream>>>(
        proj, cw[d], cb[d], convact);

    long dtot = (long)MROWS * NHEADS;
    k_dt_dA<<<(unsigned)((dtot + 255) / 256), blk, 0, stream>>>(
        proj, db[d], Al[d], dtv, dAv);

    k_scan<<<BATCH * NHEADS, blk, 0, stream>>>(convact, proj, dtv, dAv, Dh[d], ybf);

    dim3 g2(D_MODEL / 128, MROWS / 128);
    if (d == 0)
      k_gemm_bf16<0,0,0,0><<<g2, blk, 0, stream>>>(ybf, WoT, xm,
          MROWS, D_MODEL, D_INNER, SEQLEN, nullptr, nullptr);
    else
      k_gemm_bf16<0,1,1,0><<<g2, blk, 0, stream>>>(ybf, WoT, xm,
          MROWS, D_MODEL, D_INNER, SEQLEN, nullptr, nullptr);
  }

  cvt(xm, xmbf, (long)MROWS * D_MODEL);
  dim3 g3(D_MODEL / 128, MROWS / 128);
  k_gemm_bf16<0,0,0,1><<<g3, blk, 0, stream>>>(xmbf, WmT, out,
      MROWS, D_MODEL, D_MODEL, SEQLEN, xm, bm);
}
